// LinePromptedMultiScaleResampler_86260123174590
// MI455X (gfx1250) — matrix operand order repaired and checked by reading, not compiled, not run on hardware
//
#include <hip/hip_runtime.h>
#include <math.h>

// ============================================================================
// CDNA5 (gfx1250) implementation of LinePromptedMultiScaleResampler.
// Heavy matmuls run on v_wmma_f32_16x16x32_bf16 (f32 accumulate).
// Two GEMM kernels:
//   - wmma_gemm_bf16_kernel: bf16 operands in global memory; interior tiles
//     are staged with GLOBAL_LOAD_ASYNC_TO_LDS_B128 (ASYNCcnt, double-buffered
//     LDS) when the toolchain exposes the builtin, else unguarded b128 loads.
//   - wmma_gemm_kernel: f32 operands (converted to bf16 while staging) for
//     small / strided shapes.
// Epilogues fuse bias, SiLU, gated residual, and an optional bf16 mirror copy
// so downstream GEMMs consume bf16 without extra conversion passes.
//
// Input flattening assumption (jax pytree: dict keys sorted, lists in order):
//   0 feat0, 1 feat1, 2 feat2, 3 curves,
//   4..8   anchor_b1, anchor_b2, anchor_norm, anchor_w1, anchor_w2
//   9..14  glob_k, glob_o, glob_pos, glob_q, glob_queries, glob_v
//   15..19 line_k, line_o, line_pos, line_q, line_v
//   20     out_norm
//   21..25 prompt_b1, prompt_b2, prompt_norm, prompt_w1, prompt_w2
//   26..35 refine[0]: dbn,dwn,gb,gw,mlp_n,sa_n,wk,wo,wq,wv
//   36     scale_emb
//   37..65 scales[0]: dw_conv, layer0(12), layer1(12), out_b, out_w, pos_g, pw_conv
//   66..94 scales[1]: same
//   95..121 scales[2]: layer0(12), layer1(12), out_b, out_w, pos_g
//   122    type_emb
// ============================================================================

typedef __attribute__((ext_vector_type(16))) __bf16 bf16x16;
typedef __attribute__((ext_vector_type(8)))  __bf16 bf16x8;
typedef __attribute__((ext_vector_type(8)))  float  f32x8;
typedef __attribute__((ext_vector_type(4)))  float  f32x4;
typedef __attribute__((ext_vector_type(4)))  unsigned u32x4;
typedef __attribute__((ext_vector_type(2)))  unsigned u32x2;

#define TILE_M 128
#define TILE_N 128
#define TILE_K 32
#define LDSP   40   // padded LDS row stride in bf16 elements (80B -> conflict-free)

#if defined(__has_builtin)
#  if __has_builtin(__builtin_amdgcn_global_load_async_to_lds_b128) && \
      __has_builtin(__builtin_amdgcn_s_wait_asynccnt)
#    define HAVE_ASYNC_LDS 1
#  endif
#endif
#ifndef HAVE_ASYNC_LDS
#  define HAVE_ASYNC_LDS 0
#endif

#if HAVE_ASYNC_LDS
// Builtin signature (from clang diagnostics): arg0 = <4 x i32>* addrspace(1),
// arg1 = <4 x i32>* addrspace(3), then imm offset + imm cpol.
typedef int v4i_vec __attribute__((vector_size(16)));
typedef __attribute__((address_space(1))) v4i_vec* as1_v4i;
typedef __attribute__((address_space(3))) v4i_vec* as3_v4i;
__device__ __forceinline__ void async_b128(const void* g, void* l) {
  __builtin_amdgcn_global_load_async_to_lds_b128((as1_v4i)(g), (as3_v4i)(l), 0, 0);
}
#endif

__device__ __forceinline__ unsigned short bf16bits(float f) {
  unsigned u = __float_as_uint(f);
  return (unsigned short)((u + 0x7fffu + ((u >> 16) & 1u)) >> 16);  // RNE
}
__device__ __forceinline__ unsigned pack_bf16x2(float a, float b) {
  return (unsigned)bf16bits(a) | ((unsigned)bf16bits(b) << 16);
}

// ---- shared WMMA compute: 4x2 16x16 tiles per wave from one 128x128 block ----
__device__ __forceinline__ void wmma_compute(const __bf16* lA, const __bf16* lB,
                                             f32x8 (&acc)[4][2],
                                             int wm, int wn, int lr, int hi)
{
  bf16x16 bfrag[2];
#pragma unroll
  for (int ni = 0; ni < 2; ++ni) {
    const __bf16* p = lB + (wn * 32 + ni * 16 + lr) * LDSP + hi * 8;
    bf16x8 lo = *(const bf16x8*)p;
    bf16x8 hh = *(const bf16x8*)(p + 16);
    bfrag[ni] = __builtin_shufflevector(lo, hh, 0,1,2,3,4,5,6,7,8,9,10,11,12,13,14,15);
  }
#pragma unroll
  for (int mi = 0; mi < 4; ++mi) {
    const __bf16* p = lA + (wm * 64 + mi * 16 + lr) * LDSP + hi * 8;
    bf16x8 lo = *(const bf16x8*)p;
    bf16x8 hh = *(const bf16x8*)(p + 16);
    bf16x16 afrag = __builtin_shufflevector(lo, hh, 0,1,2,3,4,5,6,7,8,9,10,11,12,13,14,15);
#pragma unroll
    for (int ni = 0; ni < 2; ++ni)
      acc[mi][ni] = __builtin_amdgcn_wmma_f32_16x16x32_bf16(
          false, afrag, false, bfrag[ni], (short)0, acc[mi][ni], false, false);
  }
}

// ---- shared epilogue: bias / SiLU / gated residual / optional bf16 mirror ----
__device__ __forceinline__ void gemm_epilogue(
    const f32x8 (&acc)[4][2], float* Cm, long long csb, int crs,
    const float* bias, const float* resid, long long rsb, int rrs,
    const float* gate, unsigned short* c16,
    long long bM, long long bN, int M, int N, int act,
    int wm, int wn, int lr, int hi)
{
  float g = gate ? tanhf(gate[0]) : 1.f;
  float* Cb = Cm + (long long)blockIdx.z * csb;
  unsigned short* Cq = c16 ? c16 + (long long)blockIdx.z * csb : (unsigned short*)0;
  const float* Rb = resid ? resid + (long long)blockIdx.z * rsb : (const float*)0;
#pragma unroll
  for (int mi = 0; mi < 4; ++mi) {
#pragma unroll
    for (int ni = 0; ni < 2; ++ni) {
      long long col = bN + wn * 32 + ni * 16 + lr;
      if (col >= N) continue;
      float bv = bias ? bias[col] : 0.f;
#pragma unroll
      for (int r = 0; r < 8; ++r) {
        long long row = bM + wm * 64 + mi * 16 + r + 8 * hi;
        if (row >= M) continue;
        float v = acc[mi][ni][r] + bv;
        if (act == 1) v = v / (1.f + __expf(-v));     // SiLU
        if (Rb) v = Rb[row * (long long)rrs + col] + g * v;
        Cb[row * (long long)crs + col] = v;
        if (Cq) Cq[row * (long long)crs + col] = bf16bits(v);
      }
    }
  }
}

// ============================================================================
// bf16-operand GEMM: C[z][m][n] = ep( sum_k A[z][m][k] * B[z][n][k] )
// A: bf16, element A + z*asb + m*ars + k (K contiguous). Same for B.
// Interior (K%32==0, strides%8==0, 16B-aligned bases): async global->LDS with
// double buffering; rows are clamped so partial M/N tiles still qualify.
// ============================================================================
__global__ __launch_bounds__(256)
void wmma_gemm_bf16_kernel(const unsigned short* __restrict__ A, long long asb, int ars,
                           const unsigned short* __restrict__ Bm, long long bsb, int brs,
                           float* Cm, long long csb, int crs,
                           const float* __restrict__ bias,
                           const float* resid, long long rsb, int rrs,
                           const float* __restrict__ gate, unsigned short* c16,
                           int M, int N, int K, int act)
{
  __shared__ __bf16 lsA[2][TILE_M * LDSP];
  __shared__ __bf16 lsB[2][TILE_N * LDSP];

  const int tid  = threadIdx.x;
  const int lane = tid & 31;
  const int wv   = tid >> 5;
  const int wm   = wv >> 2;
  const int wn   = wv & 3;
  const int hi   = lane >> 4;
  const int lr   = lane & 15;

  const long long bM = (long long)blockIdx.y * TILE_M;
  const long long bN = (long long)blockIdx.x * TILE_N;
  const unsigned short* Ab = A  + (long long)blockIdx.z * asb;
  const unsigned short* Bb = Bm + (long long)blockIdx.z * bsb;

  // staging: 2 threads per row, 16 bf16 elements (2 x 16B) per thread
  const int rowT = tid >> 1;           // 0..127
  const int colE = (tid & 1) * 16;     // 0 or 16
  long long arow = bM + rowT; if (arow >= M) arow = M - 1;   // clamp: garbage
  long long brow = bN + rowT; if (brow >= N) brow = N - 1;   // rows never stored
  const unsigned short* aP = Ab + arow * (long long)ars + colE;
  const unsigned short* bP = Bb + brow * (long long)brs + colE;
  const int lofs = rowT * LDSP + colE;

  const bool interior = ((K & (TILE_K - 1)) == 0) && ((ars & 7) == 0) && ((brs & 7) == 0) &&
                        ((((size_t)Ab) & 15) == 0) && ((((size_t)Bb) & 15) == 0);

  f32x8 acc[4][2];
  const f32x8 zacc = {0.f,0.f,0.f,0.f,0.f,0.f,0.f,0.f};
#pragma unroll
  for (int mi = 0; mi < 4; ++mi)
#pragma unroll
    for (int ni = 0; ni < 2; ++ni) acc[mi][ni] = zacc;

#if HAVE_ASYNC_LDS
  if (interior) {
    auto issue = [&](int k0, int buf) {
      async_b128(aP + k0,     lsA[buf] + lofs);
      async_b128(aP + k0 + 8, lsA[buf] + lofs + 8);
      async_b128(bP + k0,     lsB[buf] + lofs);
      async_b128(bP + k0 + 8, lsB[buf] + lofs + 8);
    };
    issue(0, 0);
    for (int k0 = 0; k0 < K; k0 += TILE_K) {
      const int cur = (k0 >> 5) & 1;
      const int kn = k0 + TILE_K;
      if (kn < K) { issue(kn, cur ^ 1); __builtin_amdgcn_s_wait_asynccnt(4); }
      else        { __builtin_amdgcn_s_wait_asynccnt(0); }
      __syncthreads();
      wmma_compute(lsA[cur], lsB[cur], acc, wm, wn, lr, hi);
      __syncthreads();
    }
    gemm_epilogue(acc, Cm, csb, crs, bias, resid, rsb, rrs, gate, c16,
                  bM, bN, M, N, act, wm, wn, lr, hi);
    return;
  }
#endif
  // register-staged path (also handles K tails / misaligned strides)
  for (int k0 = 0; k0 < K; k0 += TILE_K) {
    if (interior) {
      u32x4 va0 = *(const u32x4*)(aP + k0);
      u32x4 va1 = *(const u32x4*)(aP + k0 + 8);
      u32x4 vb0 = *(const u32x4*)(bP + k0);
      u32x4 vb1 = *(const u32x4*)(bP + k0 + 8);
      *(u32x4*)(lsA[0] + lofs)     = va0;
      *(u32x4*)(lsA[0] + lofs + 8) = va1;
      *(u32x4*)(lsB[0] + lofs)     = vb0;
      *(u32x4*)(lsB[0] + lofs + 8) = vb1;
    } else {
      unsigned* la = (unsigned*)(lsA[0]) + (lofs >> 1);
      unsigned* lb = (unsigned*)(lsB[0]) + (lofs >> 1);
#pragma unroll
      for (int j = 0; j < 8; ++j) {
        int k1 = k0 + colE + j * 2;
        unsigned a0 = (k1     < K) ? aP[k0 + j*2]     : 0u;
        unsigned a1 = (k1 + 1 < K) ? aP[k0 + j*2 + 1] : 0u;
        unsigned b0 = (k1     < K) ? bP[k0 + j*2]     : 0u;
        unsigned b1 = (k1 + 1 < K) ? bP[k0 + j*2 + 1] : 0u;
        la[j] = a0 | (a1 << 16);
        lb[j] = b0 | (b1 << 16);
      }
    }
    __syncthreads();
    wmma_compute(lsA[0], lsB[0], acc, wm, wn, lr, hi);
    __syncthreads();
  }
  gemm_epilogue(acc, Cm, csb, crs, bias, resid, rsb, rrs, gate, c16,
                bM, bN, M, N, act, wm, wn, lr, hi);
}

// ============================================================================
// f32-operand GEMM (small / strided shapes); converts to bf16 while staging.
// ============================================================================
__device__ __forceinline__ void fetch_tile_f32(f32x4 (&r)[4],
    const float* const (&rowp)[4], const bool (&rOk)[4],
    int colBase, int k0, int K, int cs, bool fast)
{
  if (fast && (k0 + TILE_K <= K)) {
#pragma unroll
    for (int it = 0; it < 4; ++it)
      r[it] = *(const f32x4*)(rowp[it] + k0 + colBase);
  } else {
#pragma unroll
    for (int it = 0; it < 4; ++it) {
#pragma unroll
      for (int j = 0; j < 4; ++j) {
        int kk = k0 + colBase + j;
        r[it][j] = (rOk[it] && kk < K) ? rowp[it][(long long)kk * cs] : 0.f;
      }
    }
  }
}
__device__ __forceinline__ void store_tile_f32(const f32x4 (&r)[4],
    unsigned* __restrict__ ls32, const int (&lofs)[4])
{
#pragma unroll
  for (int it = 0; it < 4; ++it) {
    u32x2 u;
    u.x = pack_bf16x2(r[it][0], r[it][1]);
    u.y = pack_bf16x2(r[it][2], r[it][3]);
    *(u32x2*)(ls32 + lofs[it]) = u;
  }
}

__global__ __launch_bounds__(256)
void wmma_gemm_kernel(const float* __restrict__ A, long long asb, int ars, int acs,
                      const float* __restrict__ Bm, long long bsb, int brs, int bcs,
                      float* Cm, long long csb, int crs,
                      const float* __restrict__ bias,
                      const float* resid, long long rsb, int rrs,
                      const float* __restrict__ gate, unsigned short* c16,
                      int M, int N, int K, int act)
{
  __shared__ __bf16 lsA[TILE_M * LDSP];
  __shared__ __bf16 lsB[TILE_N * LDSP];
  unsigned* lsA32 = (unsigned*)lsA;
  unsigned* lsB32 = (unsigned*)lsB;

  const int tid  = threadIdx.x;
  const int lane = tid & 31;
  const int wv   = tid >> 5;
  const int wm   = wv >> 2;
  const int wn   = wv & 3;
  const int hi   = lane >> 4;
  const int lr   = lane & 15;

  const long long bM = (long long)blockIdx.y * TILE_M;
  const long long bN = (long long)blockIdx.x * TILE_N;
  const float* Ab = A  + (long long)blockIdx.z * asb;
  const float* Bb = Bm + (long long)blockIdx.z * bsb;

  const int rowT    = tid >> 3;
  const int colBase = (tid & 7) * 4;
  const float* aRow[4]; bool aOk[4]; int lofsA[4];
  const float* bRow[4]; bool bOk[4]; int lofsB[4];
#pragma unroll
  for (int it = 0; it < 4; ++it) {
    long long ra = bM + rowT + it * 32;
    long long rb = bN + rowT + it * 32;
    aOk[it] = ra < M;
    bOk[it] = rb < N;
    aRow[it] = Ab + ra * (long long)ars;
    bRow[it] = Bb + rb * (long long)brs;
    lofsA[it] = ((rowT + it * 32) * LDSP + colBase) >> 1;
    lofsB[it] = lofsA[it];
  }
  const bool fastA = (acs == 1) && ((ars & 3) == 0) &&
                     (bM + TILE_M <= M) && ((((size_t)Ab) & 15) == 0);
  const bool fastB = (bcs == 1) && ((brs & 3) == 0) &&
                     (bN + TILE_N <= N) && ((((size_t)Bb) & 15) == 0);

  f32x8 acc[4][2];
  const f32x8 zacc = {0.f,0.f,0.f,0.f,0.f,0.f,0.f,0.f};
#pragma unroll
  for (int mi = 0; mi < 4; ++mi)
#pragma unroll
    for (int ni = 0; ni < 2; ++ni) acc[mi][ni] = zacc;

  f32x4 ar[4], br[4];
  fetch_tile_f32(ar, aRow, aOk, colBase, 0, K, acs, fastA);
  fetch_tile_f32(br, bRow, bOk, colBase, 0, K, bcs, fastB);

  for (int k0 = 0; k0 < K; k0 += TILE_K) {
    store_tile_f32(ar, lsA32, lofsA);
    store_tile_f32(br, lsB32, lofsB);
    __syncthreads();
    const int kn = k0 + TILE_K;
    if (kn < K) {
      fetch_tile_f32(ar, aRow, aOk, colBase, kn, K, acs, fastA);
      fetch_tile_f32(br, bRow, bOk, colBase, kn, K, bcs, fastB);
    }
    wmma_compute(lsA, lsB, acc, wm, wn, lr, hi);
    __syncthreads();
  }
  gemm_epilogue(acc, Cm, csb, crs, bias, resid, rsb, rrs, gate, c16,
                bM, bN, M, N, act, wm, wn, lr, hi);
}

// ---------------------------------------------------------------------------
__global__ __launch_bounds__(256)
void rms_kernel(const float* __restrict__ x, float* y, unsigned short* y16,
                const float* __restrict__ g, int C)
{
  __shared__ float red[256];
  long long row = blockIdx.x;
  const float* xp = x + row * (long long)C;
  float* yp = y + row * (long long)C;
  unsigned short* yq = y16 ? y16 + row * (long long)C : (unsigned short*)0;
  int tid = threadIdx.x;
  float s = 0.f;
  for (int i = tid; i < C; i += 256) { float v = xp[i]; s += v * v; }
  red[tid] = s; __syncthreads();
  for (int w = 128; w > 0; w >>= 1) { if (tid < w) red[tid] += red[tid + w]; __syncthreads(); }
  float rr = rsqrtf(red[0] / (float)C + 1e-5f);
  for (int i = tid; i < C; i += 256) {
    float o = xp[i] * rr * g[i];
    yp[i] = o;
    if (yq) yq[i] = bf16bits(o);
  }
}

// Single-read softmax over rows of length <= 4608; optional bf16 mirror.
__global__ __launch_bounds__(256)
void softmax_kernel(float* sbuf, unsigned short* o16, const float* __restrict__ bias,
                    int len, float scale)
{
  __shared__ float red[256];
  long long row = blockIdx.x;
  float* rp = sbuf + row * (long long)len;
  unsigned short* op = o16 ? o16 + row * (long long)len : (unsigned short*)0;
  const float* bp = bias ? bias + row * (long long)len : (const float*)0;
  int tid = threadIdx.x;
  float v[18];
  float mx = -3.0e38f;
#pragma unroll
  for (int j = 0; j < 18; ++j) {
    int i = tid + j * 256;
    if (i < len) { float t = rp[i] * scale + (bp ? bp[i] : 0.f); v[j] = t; mx = fmaxf(mx, t); }
  }
  red[tid] = mx; __syncthreads();
  for (int w = 128; w > 0; w >>= 1) { if (tid < w) red[tid] = fmaxf(red[tid], red[tid + w]); __syncthreads(); }
  mx = red[0]; __syncthreads();
  float sum = 0.f;
#pragma unroll
  for (int j = 0; j < 18; ++j) {
    int i = tid + j * 256;
    if (i < len) { v[j] = __expf(v[j] - mx); sum += v[j]; }
  }
  red[tid] = sum; __syncthreads();
  for (int w = 128; w > 0; w >>= 1) { if (tid < w) red[tid] += red[tid + w]; __syncthreads(); }
  float inv = 1.f / red[0];
#pragma unroll
  for (int j = 0; j < 18; ++j) {
    int i = tid + j * 256;
    if (i < len) {
      float o = v[j] * inv;
      rp[i] = o;
      if (op) op[i] = bf16bits(o);
    }
  }
}

// flat f32 -> bf16
__global__ void cvt_kernel(unsigned short* __restrict__ out, const float* __restrict__ in,
                           long long total)
{
  long long idx = (long long)blockIdx.x * 256 + threadIdx.x;
  if (idx < total) out[idx] = bf16bits(in[idx]);
}

// gather-transpose + cvt: out[((z2*Nd + nn)*Kd) + kk] = bf16(in[z2*s2 + kk*irs + nn])
__global__ void transcvt_kernel(unsigned short* __restrict__ out, const float* __restrict__ in,
                                long long s2, int Kd, int irs, int Nd, long long total)
{
  long long idx = (long long)blockIdx.x * 256 + threadIdx.x;
  if (idx >= total) return;
  int kk = (int)(idx % Kd);
  int nn = (int)((idx / Kd) % Nd);
  long long z2 = idx / ((long long)Kd * Nd);
  out[idx] = bf16bits(in[z2 * s2 + (long long)kk * irs + nn]);
}

// Depthwise s x s, stride s, VALID; out is always (B,768,48,32).
__global__ void dwconv_kernel(const float* __restrict__ x, const float* __restrict__ w,
                              float* __restrict__ y, int H, int W, int s, long long total)
{
  long long idx = (long long)blockIdx.x * 256 + threadIdx.x;
  if (idx >= total) return;
  int ox = (int)(idx & 31);
  int oy = (int)((idx >> 5) % 48);
  int ch = (int)((idx / 1536) % 768);
  int b  = (int)(idx / (1536LL * 768));
  const float* xp = x + (((long long)b * 768 + ch) * H + (long long)oy * s) * W + (long long)ox * s;
  const float* wp = w + (long long)ch * s * s;
  float acc = 0.f;
  for (int i = 0; i < s; ++i)
    for (int j = 0; j < s; ++j)
      acc += xp[(long long)i * W + j] * wp[i * s + j];
  y[idx] = acc;
}

// (B, C, P) -> (B, P, C), f32 + optional bf16 mirror
__global__ __launch_bounds__(256)
void transpose_kernel(const float* __restrict__ x, float* __restrict__ y,
                      unsigned short* y16, int C, int P)
{
  __shared__ float tile[32][33];
  int b  = blockIdx.z;
  int p0 = blockIdx.x * 32;
  int c0 = blockIdx.y * 32;
  int tx = threadIdx.x & 31;
  int ty = threadIdx.x >> 5;
#pragma unroll
  for (int r = 0; r < 4; ++r) {
    int cc = c0 + ty + r * 8, pp = p0 + tx;
    if (cc < C && pp < P) tile[ty + r * 8][tx] = x[((long long)b * C + cc) * P + pp];
  }
  __syncthreads();
#pragma unroll
  for (int r = 0; r < 4; ++r) {
    int pp = p0 + ty + r * 8, cc = c0 + tx;
    if (pp < P && cc < C) {
      float v = tile[tx][ty + r * 8];
      long long o = ((long long)b * P + pp) * C + cc;
      y[o] = v;
      if (y16) y16[o] = bf16bits(v);
    }
  }
}

// mem[b, row0+p, :] += fourier_pe(coords(p)) + scale_emb ; D=1024, D/2=512
__global__ void addpe_kernel(float* __restrict__ mem, const float* __restrict__ posg,
                             const float* __restrict__ semb, int row0, long long total)
{
  long long idx = (long long)blockIdx.x * 256 + threadIdx.x;
  if (idx >= total) return;
  int dd = (int)(idx % 512);
  int p  = (int)((idx / 512) % 1536);
  int b  = (int)(idx / (512LL * 1536));
  float cx = ((p & 31) + 0.5f) / 32.f;
  float cy = ((p >> 5) + 0.5f) / 48.f;
  const float TWO_PI = 6.283185307179586f;
  float proj = ((2.f * cx - 1.f) * posg[dd] + (2.f * cy - 1.f) * posg[512 + dd]) * TWO_PI;
  long long base = ((long long)b * 4608 + row0 + p) * 1024;
  mem[base + dd]       += sinf(proj) + semb[dd];
  mem[base + 512 + dd] += cosf(proj) + semb[512 + dd];
}

// Bezier geometry: pts/tan (B,32,2), misc[b]={len,hei,su,sv}, pf (B,6), af (B,32,9)
__global__ void curve_kernel(const float* __restrict__ curves,
                             float* __restrict__ pts, float* __restrict__ tanv,
                             float* __restrict__ misc, float* __restrict__ pf,
                             float* __restrict__ af)
{
  int t = threadIdx.x;        // launched with exactly 128 threads
  int b = t >> 5, l = t & 31;
  const float* c = curves + b * 8;
  float p0x=c[0],p0y=c[1],p1x=c[2],p1y=c[3],p2x=c[4],p2y=c[5],p3x=c[6],p3y=c[7];
  float tt = (float)l / 31.f, om = 1.f - tt;
  float w0 = om*om*om, w1 = 3.f*om*om*tt, w2 = 3.f*om*tt*tt, w3 = tt*tt*tt;
  float px = w0*p0x + w1*p1x + w2*p2x + w3*p3x;
  float py = w0*p0y + w1*p1y + w2*p2y + w3*p3y;
  float d0 = 3.f*om*om, d1 = 6.f*om*tt, d2 = 3.f*tt*tt;
  float dx = d0*(p1x-p0x) + d1*(p2x-p1x) + d2*(p3x-p2x);
  float dy = d0*(p1y-p0y) + d1*(p2y-p1y) + d2*(p3y-p2y);
  float dn = fmaxf(sqrtf(dx*dx + dy*dy), 1e-6f);
  float tx = dx/dn, ty = dy/dn;
  pts [(b*32+l)*2+0] = px; pts [(b*32+l)*2+1] = py;
  tanv[(b*32+l)*2+0] = tx; tanv[(b*32+l)*2+1] = ty;
  __syncthreads();
  if (l == 0) {
    float L = 0.f;
    for (int i = 0; i < 31; ++i) {
      float ax = pts[(b*32+i+1)*2]   - pts[(b*32+i)*2];
      float ay = pts[(b*32+i+1)*2+1] - pts[(b*32+i)*2+1];
      L += sqrtf(ax*ax + ay*ay);
    }
    L = fmaxf(L, 0.001f);
    float ymax = fmaxf(fmaxf(p0y,p1y), fmaxf(p2y,p3y));
    float ymin = fminf(fminf(p0y,p1y), fminf(p2y,p3y));
    float Hh = fmaxf(ymax - ymin, 1.f/256.f);
    misc[b*4+0]=L; misc[b*4+1]=Hh; misc[b*4+2]=1.5f*L/32.f; misc[b*4+3]=0.5f*Hh;
    pf[b*6+0]=pts[(b*32)*2];    pf[b*6+1]=pts[(b*32)*2+1];
    pf[b*6+2]=pts[(b*32+31)*2]; pf[b*6+3]=pts[(b*32+31)*2+1];
    pf[b*6+4]=L; pf[b*6+5]=Hh;
  }
  __syncthreads();
  float L = misc[b*4], Hh = misc[b*4+1], su = misc[b*4+2], sv = misc[b*4+3];
  float* a = af + (b*32+l)*9;
  a[0]=px; a[1]=py; a[2]=tx; a[3]=ty; a[4]=tt; a[5]=L; a[6]=Hh; a[7]=su; a[8]=sv;
}

// bias[b,l,n] = -0.5*((uu/su)^2 + (vv/sv)^2)
__global__ void biasf_kernel(float* __restrict__ bias, const float* __restrict__ pts,
                             const float* __restrict__ tanv, const float* __restrict__ misc,
                             long long total)
{
  long long idx = (long long)blockIdx.x * 256 + threadIdx.x;
  if (idx >= total) return;
  int n = (int)(idx % 4608);
  int l = (int)((idx / 4608) % 32);
  int b = (int)(idx / (4608LL * 32));
  int nl = n % 1536;
  float cx = ((nl & 31) + 0.5f) / 32.f;
  float cy = ((nl >> 5) + 0.5f) / 48.f;
  float px = pts [(b*32+l)*2], py = pts [(b*32+l)*2+1];
  float tx = tanv[(b*32+l)*2], ty = tanv[(b*32+l)*2+1];
  float dx = cx - px, dy = cy - py;
  float uu = dx*tx + dy*ty;
  float vv = dx*(-ty) + dy*tx;
  float a = uu / misc[b*4+2], v = vv / misc[b*4+3];
  bias[idx] = -0.5f * (a*a + v*v);
}

__global__ void buildq_kernel(float* __restrict__ q, const float* __restrict__ anc,
                              const float* __restrict__ lpos, const float* __restrict__ temb0,
                              const float* __restrict__ prompt, long long total)
{
  long long idx = (long long)blockIdx.x * 256 + threadIdx.x;
  if (idx >= total) return;
  int d = (int)(idx % 1024);
  int l = (int)((idx / 1024) % 32);
  int b = (int)(idx / (1024LL * 32));
  q[idx] = anc[idx] + lpos[l*1024 + d] + temb0[d] + prompt[b*1024 + d];
}

__global__ void buildgq_kernel(float* __restrict__ q, const float* __restrict__ gq,
                               const float* __restrict__ gpos, const float* __restrict__ temb1,
                               long long total)
{
  long long idx = (long long)blockIdx.x * 256 + threadIdx.x;
  if (idx >= total) return;
  int d = (int)(idx % 1024);
  int g = (int)((idx / 1024) % 8);
  q[idx] = gq[g*1024 + d] + gpos[g*1024 + d] + temb1[d];
}

// ============================ host orchestration ============================

struct LayerP {
  const float *dbn,*dwn,*gb,*gw,*mlp_g,*mlp_n,*sa_g,*sa_n,*wk,*wo,*wq,*wv;
};

static LayerP gated_layer(void* const* din, int b) {
  LayerP p;
  p.dbn=(const float*)din[b+0];  p.dwn=(const float*)din[b+1];
  p.gb =(const float*)din[b+2];  p.gw =(const float*)din[b+3];
  p.mlp_g=(const float*)din[b+4]; p.mlp_n=(const float*)din[b+5];
  p.sa_g=(const float*)din[b+6];  p.sa_n=(const float*)din[b+7];
  p.wk=(const float*)din[b+8];  p.wo=(const float*)din[b+9];
  p.wq=(const float*)din[b+10]; p.wv=(const float*)din[b+11];
  return p;
}
static LayerP plain_layer(void* const* din, int b) {
  LayerP p;
  p.dbn=(const float*)din[b+0]; p.dwn=(const float*)din[b+1];
  p.gb =(const float*)din[b+2]; p.gw =(const float*)din[b+3];
  p.mlp_g=nullptr; p.mlp_n=(const float*)din[b+4];
  p.sa_g=nullptr;  p.sa_n=(const float*)din[b+5];
  p.wk=(const float*)din[b+6]; p.wo=(const float*)din[b+7];
  p.wq=(const float*)din[b+8]; p.wv=(const float*)din[b+9];
  return p;
}

static inline void gemm(hipStream_t st,
    const float* A, long long asb, int ars, int acs,
    const float* B, long long bsb, int brs, int bcs,
    float* C, long long csb, int crs,
    const float* bias, const float* resid, long long rsb, int rrs,
    const float* gate, int M, int N, int K, int act, int batch,
    unsigned short* c16 = nullptr)
{
  dim3 grid((N + TILE_N - 1) / TILE_N, (M + TILE_M - 1) / TILE_M, batch);
  wmma_gemm_kernel<<<grid, 256, 0, st>>>(A, asb, ars, acs, B, bsb, brs, bcs,
                                         C, csb, crs, bias, resid, rsb, rrs, gate, c16,
                                         M, N, K, act);
}

static inline void gemm16(hipStream_t st,
    const unsigned short* A, long long asb, int ars,
    const unsigned short* B, long long bsb, int brs,
    float* C, long long csb, int crs,
    const float* bias, const float* resid, long long rsb, int rrs,
    const float* gate, unsigned short* c16,
    int M, int N, int K, int act, int batch)
{
  dim3 grid((N + TILE_N - 1) / TILE_N, (M + TILE_M - 1) / TILE_M, batch);
  wmma_gemm_bf16_kernel<<<grid, 256, 0, st>>>(A, asb, ars, B, bsb, brs,
                                              C, csb, crs, bias, resid, rsb, rrs, gate, c16,
                                              M, N, K, act);
}

static inline void cvtb(hipStream_t st, unsigned short* dst, const float* src, long long total) {
  cvt_kernel<<<(unsigned)((total + 255) / 256), 256, 0, st>>>(dst, src, total);
}
static inline void rms(hipStream_t st, const float* x, float* y, const float* g,
                       int C, int rows, unsigned short* y16 = nullptr) {
  rms_kernel<<<rows, 256, 0, st>>>(x, y, y16, g, C);
}
static inline void softmaxk(hipStream_t st, float* s, int rows, int len, float scale,
                            const float* bias = nullptr, unsigned short* o16 = nullptr) {
  softmax_kernel<<<rows, 256, 0, st>>>(s, o16, bias, len, scale);
}

// Scale-layer transformer block, bf16 GEMM path (n=1536, c=768, nh=12).
static void run_layer16(hipStream_t st,
    float* X, unsigned short* X16, float* Nb, unsigned short* NB16,
    float* Qb, unsigned short* Q16, float* Kb, unsigned short* K16,
    float* Vb, unsigned short* VT16, float* AOb, unsigned short* AO16,
    float* Hb, unsigned short* H16, float* Sb, unsigned short* P16,
    unsigned short* W16, const LayerP& p, int Bt, int n, int c, int nh)
{
  const int rows = Bt * n;
  const int hd = c / nh;
  const int hdim = 4 * c;
  const float sc = 1.0f / sqrtf((float)hd);
  rms(st, X, Nb, p.sa_n, c, rows, NB16);
  cvtb(st, W16, p.wq, (long long)c*c);
  gemm16(st, NB16,0,c, W16,0,c, Qb,0,c, nullptr, nullptr,0,0, nullptr, Q16, rows, c, c, 0, 1);
  cvtb(st, W16, p.wk, (long long)c*c);
  gemm16(st, NB16,0,c, W16,0,c, Kb,0,c, nullptr, nullptr,0,0, nullptr, K16, rows, c, c, 0, 1);
  cvtb(st, W16, p.wv, (long long)c*c);
  gemm16(st, NB16,0,c, W16,0,c, Vb,0,c, nullptr, nullptr,0,0, nullptr, nullptr, rows, c, c, 0, 1);
  for (int bb = 0; bb < Bt; ++bb) {
    // V (n, nh*hd) -> VT16 (nh, hd, n) for K-contiguous A@V
    long long vtot = (long long)nh * hd * n;
    transcvt_kernel<<<(unsigned)((vtot + 255) / 256), 256, 0, st>>>(
        VT16, Vb + (long long)bb*n*c, hd, n, c, hd, vtot);
    gemm16(st, Q16 + (long long)bb*n*c, hd, c, K16 + (long long)bb*n*c, hd, c,
           Sb, (long long)n*n, n, nullptr, nullptr,0,0, nullptr, nullptr,
           n, n, hd, 0, nh);
    softmaxk(st, Sb, nh*n, n, sc, nullptr, P16);
    gemm16(st, P16, (long long)n*n, n, VT16, (long long)hd*n, n,
           AOb + (long long)bb*n*c, hd, c, nullptr, nullptr,0,0, nullptr,
           AO16 + (long long)bb*n*c, n, hd, n, 0, nh);
  }
  cvtb(st, W16, p.wo, (long long)c*c);
  gemm16(st, AO16,0,c, W16,0,c, X,0,c, nullptr, X,0,c, p.sa_g, nullptr, rows, c, c, 0, 1);
  rms(st, X, Nb, p.mlp_n, c, rows, NB16);
  cvtb(st, W16, p.gw, (long long)hdim*c);
  gemm16(st, NB16,0,c, W16,0,c, Hb,0,hdim, p.gb, nullptr,0,0, nullptr, H16, rows, hdim, c, 1, 1);
  cvtb(st, W16, p.dwn, (long long)c*hdim);
  gemm16(st, H16,0,hdim, W16,0,hdim, X,0,c, p.dbn, X,0,c, p.mlp_g, X16, rows, c, hdim, 0, 1);
}

// Small transformer block on the f32 path (refine: n=40, c=1024, nh=16).
static void run_layer_f32(hipStream_t st, float* X, float* Nb, float* Qb, float* Kb,
                          float* Vb, float* AOb, float* Hb, float* Sb,
                          const LayerP& p, int Bt, int n, int c, int nh)
{
  const int rows = Bt * n;
  const int hd = c / nh;
  const int hdim = 4 * c;
  const float sc = 1.0f / sqrtf((float)hd);
  rms(st, X, Nb, p.sa_n, c, rows);
  gemm(st, Nb,0,c,1, p.wq,0,c,1, Qb,0,c, nullptr, nullptr,0,0, nullptr, rows, c, c, 0, 1);
  gemm(st, Nb,0,c,1, p.wk,0,c,1, Kb,0,c, nullptr, nullptr,0,0, nullptr, rows, c, c, 0, 1);
  gemm(st, Nb,0,c,1, p.wv,0,c,1, Vb,0,c, nullptr, nullptr,0,0, nullptr, rows, c, c, 0, 1);
  for (int bb = 0; bb < Bt; ++bb) {
    const float* qb = Qb + (long long)bb*n*c;
    const float* kb = Kb + (long long)bb*n*c;
    const float* vb = Vb + (long long)bb*n*c;
    float* ob = AOb + (long long)bb*n*c;
    gemm(st, qb, hd, c, 1, kb, hd, c, 1, Sb, (long long)n*n, n,
         nullptr, nullptr,0,0, nullptr, n, n, hd, 0, nh);
    softmaxk(st, Sb, nh*n, n, sc);
    gemm(st, Sb, (long long)n*n, n, 1, vb, hd, 1, c, ob, hd, c,
         nullptr, nullptr,0,0, nullptr, n, hd, n, 0, nh);
  }
  gemm(st, AOb,0,c,1, p.wo,0,c,1, X,0,c, nullptr, X,0,c, p.sa_g, rows, c, c, 0, 1);
  rms(st, X, Nb, p.mlp_n, c, rows);
  gemm(st, Nb,0,c,1, p.gw,0,c,1, Hb,0,hdim, p.gb, nullptr,0,0, nullptr, rows, hdim, c, 1, 1);
  gemm(st, Hb,0,hdim,1, p.dwn,0,hdim,1, X,0,c, p.dbn, X,0,c, p.mlp_g, rows, c, hdim, 0, 1);
}

extern "C" void kernel_launch(void* const* d_in, const int* in_sizes, int n_in,
                              void* d_out, int out_size, void* d_ws, size_t ws_size,
                              hipStream_t stream)
{
  (void)in_sizes; (void)n_in; (void)out_size; (void)ws_size;
  auto F = [&](int i) -> const float* { return (const float*)d_in[i]; };

  const int Bt = 4, P = 1536, C = 768, D = 1024, NM = 4608;
  const float SCORE_SCALE = 0.03125f; // 1024^-0.5

  // ---- workspace layout ----
  char* base = (char*)d_ws; size_t off = 0;
  auto allocf = [&](size_t elems) -> float* {
    float* p = (float*)(base + off);
    off += ((elems * sizeof(float)) + 255) & ~(size_t)255;
    return p;
  };
  auto alloch = [&](size_t elems) -> unsigned short* {
    unsigned short* p = (unsigned short*)(base + off);
    off += ((elems * 2) + 255) & ~(size_t)255;
    return p;
  };
  float* X   = allocf((size_t)Bt*P*C);
  float* Nb  = allocf((size_t)Bt*P*C);
  float* Qb  = allocf((size_t)Bt*P*C);
  float* Kb  = allocf((size_t)Bt*P*C);
  float* Vb  = allocf((size_t)Bt*P*C);
  float* AOb = allocf((size_t)Bt*P*C);
  float* Hb  = allocf((size_t)Bt*P*3072);   // == Bt*NM*D, reused as mem@V buffer
  float* Sb  = allocf((size_t)12*P*P);      // >= Bt*NM*D, reused as mem@K buffer
  float* MEM = allocf((size_t)Bt*NM*D);
  float* DW  = allocf((size_t)Bt*C*48*32);
  float* PTS = allocf(Bt*32*2);
  float* TAN = allocf(Bt*32*2);
  float* MISC= allocf(Bt*4);
  float* PF  = allocf(Bt*6);
  float* AF  = allocf(Bt*32*9);
  float* T1  = allocf(Bt*32*D);
  float* PROMPT = allocf(Bt*D);
  float* ANCHOR = allocf(Bt*32*D);
  float* QLINE  = allocf(Bt*32*D);
  float* QP     = allocf(Bt*32*D);
  float* LSC    = allocf((size_t)Bt*32*NM);
  float* BIASF  = allocf((size_t)Bt*32*NM);
  float* LATT   = allocf(Bt*32*D);
  float* GQ     = allocf(Bt*8*D);
  float* GQP    = allocf(Bt*8*D);
  float* GSC    = allocf((size_t)Bt*8*NM);
  float* GATT   = allocf(Bt*8*D);
  float* OUT    = allocf(Bt*40*D);
  // bf16 mirrors / scratch
  unsigned short* X16  = alloch((size_t)Bt*P*C);
  unsigned short* NB16 = alloch((size_t)Bt*P*C);
  unsigned short* Q16  = alloch((size_t)Bt*P*C);
  unsigned short* K16  = alloch((size_t)Bt*P*C);
  unsigned short* AO16 = alloch((size_t)Bt*P*C);
  unsigned short* H16  = alloch((size_t)Bt*P*3072);
  unsigned short* P16  = alloch((size_t)12*P*P);
  unsigned short* VT16 = alloch((size_t)P*C);
  unsigned short* T16  = alloch((size_t)Bt*P*C);
  unsigned short* MEM16= alloch((size_t)Bt*NM*D);
  unsigned short* W16  = alloch((size_t)3072*768);

  // ---- parameter index tables ----
  const int P_TYPE_EMB = 122, P_SCALE_EMB = 36;
  struct ScaleIdx { int dw, pw, lay0, lay1, out_b, out_w, pos_g; };
  const ScaleIdx SC[3] = {
    {37, 65, 38, 50, 62, 63, 64},
    {66, 94, 67, 79, 91, 92, 93},
    {-1, -1, 95, 107, 119, 120, 121},
  };
  const int HH[3] = {192, 96, 48}, WW[3] = {128, 64, 32}, SS[3] = {4, 2, 1};

  // ======================= per-scale token pipelines =======================
  for (int i = 0; i < 3; ++i) {
    const float* feat = F(i);
    if (SS[i] > 1) {
      long long tot = (long long)Bt * C * 48 * 32;
      dwconv_kernel<<<(unsigned)((tot + 255) / 256), 256, 0, stream>>>(
          feat, F(SC[i].dw), DW, HH[i], WW[i], SS[i], tot);
      transpose_kernel<<<dim3(48, 24, Bt), 256, 0, stream>>>(DW, AOb, T16, C, P);
      cvtb(stream, W16, F(SC[i].pw), (long long)C*C);
      gemm16(stream, T16,0,C, W16,0,C, X,0,C, nullptr, nullptr,0,0, nullptr, nullptr,
             Bt*P, C, C, 0, 1);
    } else {
      transpose_kernel<<<dim3(48, 24, Bt), 256, 0, stream>>>(feat, X, nullptr, C, P);
    }
    LayerP L0 = gated_layer(d_in, SC[i].lay0);
    LayerP L1 = gated_layer(d_in, SC[i].lay1);
    run_layer16(stream, X, X16, Nb, NB16, Qb, Q16, Kb, K16, Vb, VT16,
                AOb, AO16, Hb, H16, Sb, P16, W16, L0, Bt, P, C, 12);
    run_layer16(stream, X, X16, Nb, NB16, Qb, Q16, Kb, K16, Vb, VT16,
                AOb, AO16, Hb, H16, Sb, P16, W16, L1, Bt, P, C, 12);
    // out projection into its mem segment (X16 produced by layer1's FFN2)
    cvtb(stream, W16, F(SC[i].out_w), (long long)D*C);
    gemm16(stream, X16, (long long)P*C, C, W16, 0, C,
           MEM + (size_t)i*P*D, (long long)NM*D, D, F(SC[i].out_b),
           nullptr,0,0, nullptr, nullptr, P, D, C, 0, Bt);
    long long tpe = (long long)Bt * P * (D/2);
    addpe_kernel<<<(unsigned)((tpe + 255) / 256), 256, 0, stream>>>(
        MEM, F(SC[i].pos_g), F(P_SCALE_EMB) + (size_t)i*D, i*P, tpe);
  }
  cvtb(stream, MEM16, MEM, (long long)Bt*NM*D);

  // ======================= curve geometry + prompts ========================
  curve_kernel<<<1, 128, 0, stream>>>(F(3), PTS, TAN, MISC, PF, AF);
  {
    long long tot = (long long)Bt * 32 * NM;
    biasf_kernel<<<(unsigned)((tot + 255) / 256), 256, 0, stream>>>(BIASF, PTS, TAN, MISC, tot);
  }
  // prompt = rms(mlp2(pf))
  gemm(stream, PF,0,6,1, F(24),0,6,1, T1,0,D, F(21), nullptr,0,0, nullptr, Bt, D, 6, 1, 1);
  gemm(stream, T1,0,D,1, F(25),0,D,1, PROMPT,0,D, F(22), nullptr,0,0, nullptr, Bt, D, D, 0, 1);
  rms(stream, PROMPT, PROMPT, F(23), D, Bt);
  // anchor = rms(mlp2(af))
  gemm(stream, AF,0,9,1, F(7),0,9,1, T1,0,D, F(4), nullptr,0,0, nullptr, Bt*32, D, 9, 1, 1);
  gemm(stream, T1,0,D,1, F(8),0,D,1, ANCHOR,0,D, F(5), nullptr,0,0, nullptr, Bt*32, D, D, 0, 1);
  rms(stream, ANCHOR, ANCHOR, F(6), D, Bt*32);
  {
    long long tot = (long long)Bt * 32 * D;
    buildq_kernel<<<(unsigned)((tot + 255) / 256), 256, 0, stream>>>(
        QLINE, ANCHOR, F(17), F(P_TYPE_EMB), PROMPT, tot);
  }

  // ============================ line attention =============================
  gemm(stream, QLINE,0,D,1, F(18),0,D,1, QP,0,D, nullptr, nullptr,0,0, nullptr, Bt*32, D, D, 0, 1);
  cvtb(stream, W16, F(15), (long long)D*D);
  gemm16(stream, MEM16,0,D, W16,0,D, Sb,0,D, nullptr, nullptr,0,0, nullptr, nullptr,
         Bt*NM, D, D, 0, 1);
  gemm(stream, QP, (long long)32*D, D, 1, Sb, (long long)NM*D, D, 1,
       LSC, (long long)32*NM, NM, nullptr, nullptr,0,0, nullptr, 32, NM, D, 0, Bt);
  softmaxk(stream, LSC, Bt*32, NM, SCORE_SCALE, BIASF);
  cvtb(stream, W16, F(19), (long long)D*D);
  gemm16(stream, MEM16,0,D, W16,0,D, Hb,0,D, nullptr, nullptr,0,0, nullptr, nullptr,
         Bt*NM, D, D, 0, 1);
  gemm(stream, LSC, (long long)32*NM, NM, 1, Hb, (long long)NM*D, 1, D,
       LATT, (long long)32*D, D, nullptr, nullptr,0,0, nullptr, 32, D, NM, 0, Bt);
  gemm(stream, LATT, (long long)32*D, D, 1, F(16), 0, D, 1,
       OUT, (long long)40*D, D, nullptr, QLINE, (long long)32*D, D, nullptr,
       32, D, D, 0, Bt);

  // =========================== global attention ============================
  {
    long long tot = (long long)Bt * 8 * D;
    buildgq_kernel<<<(unsigned)((tot + 255) / 256), 256, 0, stream>>>(
        GQ, F(13), F(11), F(P_TYPE_EMB) + D, tot);
  }
  gemm(stream, GQ,0,D,1, F(12),0,D,1, GQP,0,D, nullptr, nullptr,0,0, nullptr, Bt*8, D, D, 0, 1);
  cvtb(stream, W16, F(9), (long long)D*D);
  gemm16(stream, MEM16,0,D, W16,0,D, Sb,0,D, nullptr, nullptr,0,0, nullptr, nullptr,
         Bt*NM, D, D, 0, 1);
  gemm(stream, GQP, (long long)8*D, D, 1, Sb, (long long)NM*D, D, 1,
       GSC, (long long)8*NM, NM, nullptr, nullptr,0,0, nullptr, 8, NM, D, 0, Bt);
  softmaxk(stream, GSC, Bt*8, NM, SCORE_SCALE);
  cvtb(stream, W16, F(14), (long long)D*D);
  gemm16(stream, MEM16,0,D, W16,0,D, Hb,0,D, nullptr, nullptr,0,0, nullptr, nullptr,
         Bt*NM, D, D, 0, 1);
  gemm(stream, GSC, (long long)8*NM, NM, 1, Hb, (long long)NM*D, 1, D,
       GATT, (long long)8*D, D, nullptr, nullptr,0,0, nullptr, 8, D, NM, 0, Bt);
  gemm(stream, GATT, (long long)8*D, D, 1, F(10), 0, D, 1,
       OUT + 32*D, (long long)40*D, D, nullptr, GQ, (long long)8*D, D, nullptr,
       8, D, D, 0, Bt);

  // ============================== refine + out =============================
  LayerP R = plain_layer(d_in, 26);
  run_layer_f32(stream, OUT, Nb, Qb, Kb, Vb, AOb, Hb, Sb, R, Bt, 40, D, 16);
  rms(stream, OUT, (float*)d_out, F(20), D, Bt*40);
}
